// Attention_18562848653816
// MI455X (gfx1250) — matrix operand order repaired and checked by reading, not compiled, not run on hardware
//
#include <hip/hip_runtime.h>
#include <hip/hip_bf16.h>
#include <math.h>

// Problem dims (fixed by the reference)
#define BB   4
#define SS   1024
#define HIDN 2048
#define NH   16
#define NKV  4
#define DH   128
#define SCALE 0.08838834764831845f  // 1/sqrt(128)

typedef __attribute__((ext_vector_type(16))) __bf16 bf16x16;
typedef __attribute__((ext_vector_type(8)))  __bf16 bf16x8;
typedef __attribute__((ext_vector_type(8)))  float  f32x8;

#if defined(__has_builtin)
#if __has_builtin(__builtin_amdgcn_tensor_load_to_lds)
#define HAVE_TDM 1
#endif
#endif
#ifndef HAVE_TDM
#define HAVE_TDM 0
#endif

__device__ __forceinline__ f32x8 zero8() {
  f32x8 z;
#pragma unroll
  for (int i = 0; i < 8; ++i) z[i] = 0.0f;
  return z;
}

__device__ __forceinline__ f32x8 wmma_bf16(bf16x16 a, bf16x16 b, f32x8 c) {
  return __builtin_amdgcn_wmma_f32_16x16x32_bf16(false, a, false, b, (short)0, c,
                                                 false, false);
}

__device__ __forceinline__ bf16x8 cvt8(float4 a, float4 b) {
  bf16x8 r;
  r[0] = (__bf16)a.x; r[1] = (__bf16)a.y; r[2] = (__bf16)a.z; r[3] = (__bf16)a.w;
  r[4] = (__bf16)b.x; r[5] = (__bf16)b.y; r[6] = (__bf16)b.z; r[7] = (__bf16)b.w;
  return r;
}

#if HAVE_TDM
typedef unsigned int u32x4 __attribute__((ext_vector_type(4)));
typedef int          i32x8 __attribute__((ext_vector_type(8)));
typedef int          i32x4 __attribute__((ext_vector_type(4)));

// TDM copy of one contiguous 8KB block (1024 x 8B) global -> LDS.
// D# per CDNA5 ISA sec 8.3/8.4: group0 = {flags, lds_addr, global_addr, type},
// group1 = {data_size=8B, tensor_dim0=1024, tensor_dim1=1, tile_dim0=1024,
//           tile_dim1=1, tensor_dim0_stride=1024}; groups 2/3 unused (<=2D).
// This toolchain exposes the 6-arg builtin (extra descriptor group, zeroed).
__device__ __forceinline__ void tdm_copy_8k(const void* gsrc, unsigned lds_off) {
  unsigned long long ga = (unsigned long long)gsrc;
  u32x4 g0;
  g0[0] = 1u;                                   // count=1, user descriptor
  g0[1] = lds_off;                              // lds_addr (bytes)
  g0[2] = (unsigned)(ga & 0xFFFFFFFFu);         // global_addr[31:0]
  g0[3] = (unsigned)((ga >> 32) & 0x1FFFFFFu)   // global_addr[56:32]
          | (2u << 30);                         // type = 2 ("image")
  i32x8 g1;
  g1[0] = 0x00030000;          // workgroup_mask=0, data_size=3 (8B)
  g1[1] = (int)(1024u << 16);  // tensor_dim0[15:0]=1024
  g1[2] = (int)(1u << 16);     // tensor_dim0[31:16]=0 | tensor_dim1[15:0]=1
  g1[3] = (int)(1024u << 16);  // tensor_dim1[31:16]=0 | tile_dim0=1024
  g1[4] = 1;                   // tile_dim1=1, tile_dim2=0
  g1[5] = 1024;                // tensor_dim0_stride[31:0]
  g1[6] = 0;                   // stride hi | tensor_dim1_stride lo
  g1[7] = 0;
  i32x4 z4 = {0, 0, 0, 0};
  i32x8 z8;
#pragma unroll
  for (int i = 0; i < 8; ++i) z8[i] = 0;
  __builtin_amdgcn_tensor_load_to_lds(g0, g1, z4, z4, z8, 0);
}

__device__ __forceinline__ void tdm_wait() {
#if __has_builtin(__builtin_amdgcn_s_wait_tensorcnt)
  __builtin_amdgcn_s_wait_tensorcnt(0);
#else
  asm volatile("s_wait_tensorcnt 0x0" ::: "memory");
#endif
}
#endif  // HAVE_TDM

// ===========================================================================
// Fragment-order packing. All packed buffers hold, per 16x(K=32) fragment
// block, 32 lanes x 16 bf16 elements contiguous (32B per lane):
//  A-frag: lane=((k>>3)&1)*16+m,  e=(k>>4)*8+(k&7)
//  B-frag: lane=((k>>3)&1)*16+n,  e=(k>>4)*8+(k&7)
// ===========================================================================

// x[M,K] fp32 row-major -> A-frag order
__global__ void pack_a_kernel(const float* __restrict__ src, __bf16* __restrict__ dst) {
  int c = blockIdx.x * blockDim.x + threadIdx.x;  // chunk of 8
  int s  = c >> 8;
  int kc = c & 255;
  int k  = kc * 8;
  const float* p = src + (size_t)s * HIDN + k;
  float4 f0 = *(const float4*)p;
  float4 f1 = *(const float4*)(p + 4);
  int rb = s >> 4, m = s & 15, kb = k >> 5, k8 = (k >> 3) & 3;
  int lane = (k8 & 1) * 16 + m;
  size_t d = (((size_t)rb * (HIDN / 32) + kb) * 32 + lane) * 16 + (k8 >> 1) * 8;
  *(bf16x8*)&dst[d] = cvt8(f0, f1);
}

// W[K,N] fp32 row-major -> B-frag order
template <int N, int K>
__global__ void pack_w_kernel(const float* __restrict__ W, __bf16* __restrict__ wp) {
  int idx = blockIdx.x * blockDim.x + threadIdx.x;  // N*(K/16) threads
  int nn  = idx % N;
  int c16 = idx / N;
  int k0  = c16 * 16;
  bf16x16 v;
#pragma unroll
  for (int u = 0; u < 16; ++u) v[u] = (__bf16)W[(size_t)((c16 >> 1) * 32 + (u >> 3) * 16 + (c16 & 1) * 8 + (u & 7)) * N + nn];
  int kb = c16 >> 1, g = c16 & 1;
  size_t d = ((((size_t)(nn >> 6) * (K / 32) + kb) * 4 + ((nn >> 4) & 3)) * 512) +
             ((size_t)(g * 16 + (nn & 15)) * 16);
  *(bf16x16*)&wp[d] = v;
}

// ===========================================================================
// Frag-direct GEMM: no LDS, no syncs; per k-step: 10x b128 loads + 4 WMMA.
// ===========================================================================
template <int N, int K>
__global__ __launch_bounds__(128) void gemm_frag_kernel(const __bf16* __restrict__ Ap,
                                                        const __bf16* __restrict__ Bp,
                                                        float* __restrict__ C) {
  const int tid  = threadIdx.x;
  const int wave = tid >> 5;
  const int lane = tid & 31;
  const int g    = lane >> 4;
  const int n    = lane & 15;
  const int m0   = blockIdx.y * 64;
  const int n0   = blockIdx.x * 64;
  const int rb   = (m0 >> 4) + wave;

  f32x8 acc[4];
#pragma unroll
  for (int t = 0; t < 4; ++t) acc[t] = zero8();

  const __bf16* apBase = Ap + ((size_t)rb * (K / 32)) * 512 + lane * 16;
  const __bf16* bpBase = Bp + ((size_t)(n0 >> 6) * (K / 32)) * 4 * 512 + lane * 16;

#pragma unroll 4
  for (int kb = 0; kb < K / 32; ++kb) {
    bf16x16 af = *(const bf16x16*)(apBase + (size_t)kb * 512);
#pragma unroll
    for (int t = 0; t < 4; ++t) {
      bf16x16 bfr = *(const bf16x16*)(bpBase + ((size_t)kb * 4 + t) * 512);
      acc[t] = wmma_bf16(af, bfr, acc[t]);
    }
  }

#pragma unroll
  for (int t = 0; t < 4; ++t)
#pragma unroll
    for (int r = 0; r < 8; ++r)
      C[(size_t)(m0 + wave * 16 + g * 8 + r) * N + n0 + t * 16 + n] = acc[t][r];
}

// ===========================================================================
// RoPE + pack q -> A-frag order per (b,h); scale folded.
// ===========================================================================
__global__ void rope_pack_q_kernel(const float* __restrict__ q,
                                   const float* __restrict__ cosb,
                                   const float* __restrict__ sinb,
                                   __bf16* __restrict__ qp) {
  int idx = blockIdx.x * blockDim.x + threadIdx.x;  // BB*SS*NH*64
  int d   = idx & 63;
  int h   = (idx >> 6) & (NH - 1);
  int row = idx >> 10;
  int s   = row & (SS - 1);
  int b   = row >> 10;
  const float* base = q + (size_t)row * (NH * DH) + h * DH;
  float x1 = base[d], x2 = base[d + 64];
  float c  = cosb[s * DH + d], sn = sinb[s * DH + d];
  float o1 = (x1 * c - x2 * sn) * SCALE;
  float o2 = (x2 * c + x1 * sn) * SCALE;
  int m = s & 15, rb = s >> 4;
  size_t blk = (((size_t)(b * NH + h) * 64 + rb) * 4);
#pragma unroll
  for (int half = 0; half < 2; ++half) {
    int dv = d + half * 64;
    float val = half ? o2 : o1;
    int ks = dv >> 5, dd = dv & 31;
    int lane = ((dd >> 3) & 1) * 16 + m;
    int e = ((dd >> 4) & 1) * 8 + (dd & 7);
    qp[(blk + ks) * 512 + lane * 16 + e] = (__bf16)val;
  }
}

// RoPE + pack k -> score-B-frag order per (b,kvh,32-key block)
__global__ void rope_pack_k_kernel(const float* __restrict__ k,
                                   const float* __restrict__ cosb,
                                   const float* __restrict__ sinb,
                                   __bf16* __restrict__ kp) {
  int idx = blockIdx.x * blockDim.x + threadIdx.x;  // BB*SS*NKV*64
  int d   = idx & 63;
  int kvh = (idx >> 6) & (NKV - 1);
  int row = idx >> 8;
  int s   = row & (SS - 1);
  int b   = row >> 10;
  const float* base = k + (size_t)row * (NKV * DH) + kvh * DH;
  float x1 = base[d], x2 = base[d + 64];
  float c  = cosb[s * DH + d], sn = sinb[s * DH + d];
  float o1 = x1 * c - x2 * sn;
  float o2 = x2 * c + x1 * sn;
  int j = s & 31, jb = s >> 5;
  int kt = j >> 4, n = j & 15;
  size_t blk = (((size_t)(b * NKV + kvh) * 32 + jb) * 8);
#pragma unroll
  for (int half = 0; half < 2; ++half) {
    int dv = d + half * 64;
    float val = half ? o2 : o1;
    int ks = dv >> 5, dd = dv & 31;
    kp[(blk + ks * 2 + kt) * 512 + ((size_t)(((dd >> 3) & 1) * 16 + n) * 16) + ((dd >> 4) & 1) * 8 + (dd & 7)] =
        (__bf16)val;
  }
}

// pack v -> PV-B-frag order per (b,kvh,32-key block)
__global__ void pack_v_kernel(const float* __restrict__ v, __bf16* __restrict__ vp) {
  int idx = blockIdx.x * blockDim.x + threadIdx.x;  // BB*NKV*32*4*128
  int d   = idx & 127;
  int jc  = (idx >> 7) & 3;
  int jb  = (idx >> 9) & 31;
  int kvh = (idx >> 14) & (NKV - 1);
  int b   = idx >> 16;
  bf16x8 r;
#pragma unroll
  for (int u = 0; u < 8; ++u)
    r[u] = (__bf16)v[((size_t)(b * SS + jb * 32 + jc * 8 + u) * NKV + kvh) * DH + d];
  size_t dst = ((((size_t)(b * NKV + kvh) * 32 + jb) * 8 + (d >> 4)) * 512) +
               ((size_t)((jc & 1) * 16 + (d & 15)) * 16) + (jc >> 1) * 8;
  *(bf16x8*)&vp[dst] = r;
}

// ===========================================================================
// Causal GQA flash attention; K/V blocks staged to LDS by TDM (tensor_load_
// to_lds + s_wait_tensorcnt) when available, else vector-copy fallback.
// ===========================================================================
__global__ __launch_bounds__(128) void attn_kernel(const __bf16* __restrict__ qp,
                                                   const __bf16* __restrict__ kp,
                                                   const __bf16* __restrict__ vp,
                                                   __bf16* __restrict__ aws) {
  __shared__ alignas(32) __bf16 Kt_sw[8 * 32 * 16];
  __shared__ alignas(32) __bf16 Vt_sw[8 * 32 * 16];
  __shared__ alignas(32) __bf16 Pscr[4 * 32 * 16];

  const int tid  = threadIdx.x;
  const int wave = tid >> 5;
  const int lane = tid & 31;
  const int g    = lane >> 4;
  const int n    = lane & 15;
  const int bh   = blockIdx.y;
  const int b    = bh / NH;
  const int h    = bh % NH;
  const int kvh  = h / (NH / NKV);
  const int qb   = blockIdx.x;
  const int q0   = qb * 64 + wave * 16;

  bf16x16 qf[4];
  {
    const __bf16* qpb =
        qp + (((size_t)(b * NH + h) * 64 + (q0 >> 4)) * 4) * 512 + lane * 16;
#pragma unroll
    for (int ks = 0; ks < 4; ++ks) qf[ks] = *(const bf16x16*)(qpb + ks * 512);
  }

  f32x8 O[8];
#pragma unroll
  for (int t = 0; t < 8; ++t) O[t] = zero8();
  float runmax[8], runsum[8];
#pragma unroll
  for (int r = 0; r < 8; ++r) { runmax[r] = -1e30f; runsum[r] = 0.0f; }

  const __bf16* kbase = kp + ((size_t)(b * NKV + kvh) * 32) * 8 * 512;
  const __bf16* vbase = vp + ((size_t)(b * NKV + kvh) * 32) * 8 * 512;

  const int njb = (qb + 1) * 2;  // uniform; masking handles causality
  for (int jb = 0; jb < njb; ++jb) {
    const __bf16* kblock = kbase + (size_t)jb * 4096;
    const __bf16* vblock = vbase + (size_t)jb * 4096;
    __syncthreads();
#if HAVE_TDM
    if (wave == 0) {
      tdm_copy_8k(kblock, (unsigned)(unsigned long long)(uintptr_t)&Kt_sw[0]);
      tdm_copy_8k(vblock, (unsigned)(unsigned long long)(uintptr_t)&Vt_sw[0]);
      tdm_wait();
    }
#else
#pragma unroll
    for (int i = 0; i < 4; ++i) {
      int o = (tid + i * 128) * 8;
      *(uint4*)&Kt_sw[o] = *(const uint4*)&kblock[o];
      *(uint4*)&Vt_sw[o] = *(const uint4*)&vblock[o];
    }
#endif
    if (jb + 1 < njb) {
      __builtin_prefetch(kblock + 4096 + tid * 32, 0, 1);
      __builtin_prefetch(vblock + 4096 + tid * 32, 0, 1);
    }
    __syncthreads();

    // Scores: two 16x16 key tiles, 4 d-steps of 32
    f32x8 s0 = zero8(), s1 = zero8();
#pragma unroll
    for (int ks = 0; ks < 4; ++ks) {
      bf16x16 bk0 = *(const bf16x16*)&Kt_sw[((ks * 2 + 0) * 32 + lane) << 4];
      bf16x16 bk1 = *(const bf16x16*)&Kt_sw[((ks * 2 + 1) * 32 + lane) << 4];
      s0 = wmma_bf16(qf[ks], bk0, s0);
      s1 = wmma_bf16(qf[ks], bk1, s1);
    }

    // Causal mask + online softmax in C-fragment layout
    float al[8];
#pragma unroll
    for (int r = 0; r < 8; ++r) {
      int qr = q0 + g * 8 + r;
      float a0 = (jb * 32 + n      <= qr) ? s0[r] : -1e30f;
      float a1 = (jb * 32 + 16 + n <= qr) ? s1[r] : -1e30f;
      float m2 = fmaxf(a0, a1);
#pragma unroll
      for (int off = 1; off < 16; off <<= 1)
        m2 = fmaxf(m2, __shfl_xor(m2, off, 32));
      float newmax = fmaxf(runmax[r], m2);
      float alpha  = __expf(runmax[r] - newmax);
      float p0 = __expf(a0 - newmax);
      float p1 = __expf(a1 - newmax);
      s0[r] = p0; s1[r] = p1;
      float sum = p0 + p1;
#pragma unroll
      for (int off = 1; off < 16; off <<= 1)
        sum += __shfl_xor(sum, off, 32);
      runsum[r] = runsum[r] * alpha + sum;
      runmax[r] = newmax;
      al[r] = alpha;
    }
#pragma unroll
    for (int t = 0; t < 8; ++t)
#pragma unroll
      for (int r = 0; r < 8; ++r) O[t][r] *= al[r];

    // P: C-layout -> A-frag LDS
    {
      __bf16* dstp = &Pscr[((wave * 32 + ((n >> 3) & 1) * 16 + g * 8) << 4)];
#pragma unroll
      for (int r = 0; r < 8; ++r) {
        dstp[(r << 4) + (n & 7)]     = (__bf16)s0[r];
        dstp[(r << 4) + 8 + (n & 7)] = (__bf16)s1[r];
      }
    }
    __syncthreads();
    bf16x16 pf = *(const bf16x16*)&Pscr[(wave * 32 + lane) << 4];
#pragma unroll
    for (int t = 0; t < 8; ++t) {
      bf16x16 bv = *(const bf16x16*)&Vt_sw[(t * 32 + lane) << 4];
      O[t] = wmma_bf16(pf, bv, O[t]);
    }
  }

  // Epilogue: normalize, transpose through LDS, write bf16 A-frag order.
  float inv[8];
#pragma unroll
  for (int r = 0; r < 8; ++r) inv[r] = 1.0f / runsum[r];
  __bf16* dstp = &Pscr[((wave * 32 + ((n >> 3) & 1) * 16 + g * 8) << 4)];
  const size_t grb = (size_t)((b * SS + q0) >> 4);
#pragma unroll
  for (int tt = 0; tt < 4; ++tt) {
    __syncthreads();
#pragma unroll
    for (int r = 0; r < 8; ++r) {
      dstp[(r << 4) + (n & 7)]     = (__bf16)(O[2 * tt][r] * inv[r]);
      dstp[(r << 4) + 8 + (n & 7)] = (__bf16)(O[2 * tt + 1][r] * inv[r]);
    }
    __syncthreads();
    bf16x16 ofr = *(const bf16x16*)&Pscr[(wave * 32 + lane) << 4];
    *(bf16x16*)&aws[((grb * (HIDN / 32) + (h * 4 + tt)) * 32 + lane) << 4] = ofr;
  }
}

// ===========================================================================
extern "C" void kernel_launch(void* const* d_in, const int* in_sizes, int n_in,
                              void* d_out, int out_size, void* d_ws, size_t ws_size,
                              hipStream_t stream) {
  (void)in_sizes; (void)n_in; (void)out_size; (void)ws_size;
  const float* x    = (const float*)d_in[0];
  const float* cosb = (const float*)d_in[1];
  const float* sinb = (const float*)d_in[2];
  const float* wq   = (const float*)d_in[3];
  const float* wk   = (const float*)d_in[4];
  const float* wv   = (const float*)d_in[5];
  const float* wo   = (const float*)d_in[6];
  float* out = (float*)d_out;

  const int M = BB * SS;  // 4096
  char* ws = (char*)d_ws;
  size_t off = 0;
  float*  qf32 = (float*)(ws + off);  off += (size_t)M * HIDN * 4;
  float*  kf32 = (float*)(ws + off);  off += (size_t)M * NKV * DH * 4;
  float*  vf32 = (float*)(ws + off);  off += (size_t)M * NKV * DH * 4;
  __bf16* xpk  = (__bf16*)(ws + off); off += (size_t)M * HIDN * 2;
  __bf16* wqp  = (__bf16*)(ws + off); off += (size_t)HIDN * HIDN * 2;
  __bf16* wkp  = (__bf16*)(ws + off); off += (size_t)HIDN * NKV * DH * 2;
  __bf16* wvp  = (__bf16*)(ws + off); off += (size_t)HIDN * NKV * DH * 2;
  __bf16* wop  = (__bf16*)(ws + off); off += (size_t)HIDN * HIDN * 2;
  __bf16* qpk  = (__bf16*)(ws + off); off += (size_t)M * NH * DH * 2;
  __bf16* kpk  = (__bf16*)(ws + off); off += (size_t)M * NKV * DH * 2;
  __bf16* vpk  = (__bf16*)(ws + off); off += (size_t)M * NKV * DH * 2;
  __bf16* awsp = (__bf16*)qf32;  // alias: qf32 dead after rope_pack_q

  dim3 blk(256);
  pack_a_kernel<<<(M * HIDN / 8) / 256, blk, 0, stream>>>(x, xpk);
  pack_w_kernel<HIDN, HIDN><<<(HIDN * (HIDN / 16)) / 256, blk, 0, stream>>>(wq, wqp);
  pack_w_kernel<NKV * DH, HIDN><<<(NKV * DH * (HIDN / 16)) / 256, blk, 0, stream>>>(wk, wkp);
  pack_w_kernel<NKV * DH, HIDN><<<(NKV * DH * (HIDN / 16)) / 256, blk, 0, stream>>>(wv, wvp);
  pack_w_kernel<HIDN, HIDN><<<(HIDN * (HIDN / 16)) / 256, blk, 0, stream>>>(wo, wop);

  gemm_frag_kernel<HIDN, HIDN><<<dim3(HIDN / 64, M / 64), 128, 0, stream>>>(xpk, wqp, qf32);
  gemm_frag_kernel<NKV * DH, HIDN><<<dim3((NKV * DH) / 64, M / 64), 128, 0, stream>>>(xpk, wkp, kf32);
  gemm_frag_kernel<NKV * DH, HIDN><<<dim3((NKV * DH) / 64, M / 64), 128, 0, stream>>>(xpk, wvp, vf32);

  rope_pack_q_kernel<<<(M * NH * 64) / 256, blk, 0, stream>>>(qf32, cosb, sinb, qpk);
  rope_pack_k_kernel<<<(M * NKV * 64) / 256, blk, 0, stream>>>(kf32, cosb, sinb, kpk);
  pack_v_kernel<<<(BB * NKV * 32 * 4 * 128) / 256, blk, 0, stream>>>(vf32, vpk);

  attn_kernel<<<dim3(SS / 64, BB * NH), 128, 0, stream>>>(qpk, kpk, vpk, awsp);

  gemm_frag_kernel<HIDN, HIDN><<<dim3(HIDN / 64, M / 64), 128, 0, stream>>>(awsp, wop, out);
}
